// TGAT_541165879485
// MI455X (gfx1250) — compile-verified, matrix operand order checked by reading, not run on hardware
//
#include <hip/hip_runtime.h>
#include <hip/hip_bf16.h>
#include <cstdint>

// ---------------------------------------------------------------------------
// TransformerConv x2 + linear head for gfx1250 (wave32, WMMA 16x16x32 f16).
// GEMMs: one wave per 16-row output stripe, A fragments register-resident,
// 32 WMMAs/wave.  Edge softmax: L2-resident gathers + native fp32 atomics.
// ---------------------------------------------------------------------------

typedef __attribute__((ext_vector_type(16))) _Float16 v16h;
typedef __attribute__((ext_vector_type(8)))  _Float16 v8h;
typedef __attribute__((ext_vector_type(8)))  float    v8f;

#define NN 100000      // nodes
#define NE 1600000     // edges
#define FD 128         // feature dim (= H*D = 2*64)
#define DH 64          // per-head dim

__device__ __forceinline__ void atomicMaxF(float* a, float v) {
    __hip_atomic_fetch_max(a, v, __ATOMIC_RELAXED, __HIP_MEMORY_SCOPE_AGENT);
}
__device__ __forceinline__ void atomicAddF(float* a, float v) {
    __hip_atomic_fetch_add(a, v, __ATOMIC_RELAXED, __HIP_MEMORY_SCOPE_AGENT);
}

// ---------------------------------------------------------------------------
// Weight packing: f32 row-major [K, Ncols] -> f16 B-fragment layout.
// B frag (16x16x32 f16, wave32): lanes 0-15 hold N=0..15 / lanes 16-31 hold
// N=0..15 with K offset +8.  VGPR v holds K pair (2v,2v+1) for v<4 and
// (16+2(v-4), ...) for v>=4, within a 32-wide K block.
// Packed entry: wp[((kk*NCT + ct)*32 + lane)*16 + e]  -> one contiguous 32B
// load per lane in the GEMM.
// mats 0..3: layer1 {Wq,Wk,Wv,Ws}; 4..7: layer2; 8: Wc padded to 48 cols.
// ---------------------------------------------------------------------------
__global__ void pack_weights_kernel(
    const float* Wq1, const float* Wk1, const float* Wv1, const float* Ws1,
    const float* Wq2, const float* Wk2, const float* Wv2, const float* Ws2,
    const float* Wc, _Float16* wpack)
{
    int mat = blockIdx.y;                                // 0..8
    int idx = blockIdx.x * blockDim.x + threadIdx.x;     // entry index
    const float* W;
    switch (mat) {
        case 0: W = Wq1; break; case 1: W = Wk1; break;
        case 2: W = Wv1; break; case 3: W = Ws1; break;
        case 4: W = Wq2; break; case 5: W = Wk2; break;
        case 6: W = Wv2; break; case 7: W = Ws2; break;
        default: W = Wc; break;
    }
    int nct   = (mat < 8) ? 8 : 3;     // column tiles
    int ncols = (mat < 8) ? 128 : 40;  // valid columns (Wc padded to 48)
    int nent  = 4 * nct * 32;
    if (idx >= nent) return;

    int lane = idx & 31;
    int ct   = (idx >> 5) % nct;
    int kk   = (idx >> 5) / nct;

    _Float16* dst = wpack + (size_t)mat * 16384 +
                    ((size_t)(kk * nct + ct) * 32 + lane) * 16;
    int base8 = (lane & 16) ? 8 : 0;
    int n     = ct * 16 + (lane & 15);

#pragma unroll
    for (int e = 0; e < 16; ++e) {
        int v  = e >> 1, hl = e & 1;
        int ki = (v < 4) ? (2 * v + hl) : (16 + 2 * (v - 4) + hl);
        int K  = kk * 32 + base8 + ki;
        float val = (n < ncols) ? W[(size_t)K * ncols + n] : 0.0f;
        dst[e] = (_Float16)val;
    }
}

// f32 -> f16 activation copy (A operand buffer)
__global__ void cvt_f16_kernel(const float* __restrict__ src,
                               _Float16* __restrict__ dst, size_t n)
{
    size_t i = (size_t)blockIdx.x * blockDim.x + threadIdx.x;
    if (i < n) dst[i] = (_Float16)src[i];
}

// ---------------------------------------------------------------------------
// Fused q/k/v/skip GEMM: out = xh[100000,128] @ W + b.
// One wave owns a full 16x128 output stripe: 4 A fragments loaded once into
// registers (two contiguous 16B loads each), then 8 column tiles x 4 WMMAs.
// blockIdx.y selects which weight/output.
// ---------------------------------------------------------------------------
__global__ void gemm_qkvs_kernel(const _Float16* __restrict__ xh,
                                 const _Float16* __restrict__ wpack4,
                                 const float* b0, const float* b1,
                                 const float* b2, const float* b3,
                                 float* o0, float* o1, float* o2, float* o3)
{
    int which = blockIdx.y;
    const _Float16* wp = wpack4 + (size_t)which * 16384;
    const float* bias  = (which == 0) ? b0 : (which == 1) ? b1 : (which == 2) ? b2 : b3;
    float* out         = (which == 0) ? o0 : (which == 1) ? o1 : (which == 2) ? o2 : o3;

    int lane   = threadIdx.x & 31;
    int wave   = threadIdx.x >> 5;
    int stripe = blockIdx.x * 8 + wave;          // 16-row output stripe
    if (stripe >= NN / 16) return;               // wave-uniform exit
    int m0   = stripe * 16;
    int row  = m0 + (lane & 15);
    int koff = (lane & 16) ? 8 : 0;

    // A fragments for K=0..127, register resident, reused by all 8 col tiles.
    v16h a[4];
#pragma unroll
    for (int kk = 0; kk < 4; ++kk) {
        const _Float16* ap = xh + (size_t)row * FD + kk * 32 + koff;
        v8h lo = *(const v8h*)(ap);
        v8h hi = *(const v8h*)(ap + 16);
#pragma unroll
        for (int i = 0; i < 8; ++i) { a[kk][i] = lo[i]; a[kk][8 + i] = hi[i]; }
    }

    int ncol  = lane & 15;
    int mbase = m0 + ((lane & 16) ? 8 : 0);

#pragma unroll
    for (int ct = 0; ct < 8; ++ct) {
        int n = ct * 16 + ncol;
        float bv = bias[n];
        v8f c;
#pragma unroll
        for (int r = 0; r < 8; ++r) c[r] = bv;
#pragma unroll
        for (int kk = 0; kk < 4; ++kk) {
            v16h b = *(const v16h*)(wp + ((size_t)(kk * 8 + ct) * 32 + lane) * 16);
            c = __builtin_amdgcn_wmma_f32_16x16x32_f16(false, a[kk], false, b,
                                                       (short)0, c, false, false);
        }
#pragma unroll
        for (int r = 0; r < 8; ++r)
            out[(size_t)(mbase + r) * FD + n] = c[r];
    }
}

// Final head: out[100000,40] = xh @ Wc + bc  (cols padded to 48, store-masked)
__global__ void gemm_head_kernel(const _Float16* __restrict__ xh,
                                 const _Float16* __restrict__ wpc,
                                 const float* __restrict__ bc,
                                 float* __restrict__ out)
{
    int lane   = threadIdx.x & 31;
    int wave   = threadIdx.x >> 5;
    int stripe = blockIdx.x * 8 + wave;
    if (stripe >= NN / 16) return;
    int m0   = stripe * 16;
    int row  = m0 + (lane & 15);
    int koff = (lane & 16) ? 8 : 0;

    v16h a[4];
#pragma unroll
    for (int kk = 0; kk < 4; ++kk) {
        const _Float16* ap = xh + (size_t)row * FD + kk * 32 + koff;
        v8h lo = *(const v8h*)(ap);
        v8h hi = *(const v8h*)(ap + 16);
#pragma unroll
        for (int i = 0; i < 8; ++i) { a[kk][i] = lo[i]; a[kk][8 + i] = hi[i]; }
    }

    int ncol  = lane & 15;
    int mbase = m0 + ((lane & 16) ? 8 : 0);

#pragma unroll
    for (int ct = 0; ct < 3; ++ct) {
        int n = ct * 16 + ncol;
        float bv = (n < 40) ? bc[n] : 0.0f;
        v8f c;
#pragma unroll
        for (int r = 0; r < 8; ++r) c[r] = bv;
#pragma unroll
        for (int kk = 0; kk < 4; ++kk) {
            v16h b = *(const v16h*)(wpc + ((size_t)(kk * 3 + ct) * 32 + lane) * 16);
            c = __builtin_amdgcn_wmma_f32_16x16x32_f16(false, a[kk], false, b,
                                                       (short)0, c, false, false);
        }
        if (n < 40) {
#pragma unroll
            for (int r = 0; r < 8; ++r)
                out[(size_t)(mbase + r) * 40 + n] = c[r];
        }
    }
}

// m = -inf, denom = 0  (2N entries each)
__global__ void init_mdenom_kernel(float* m, float* denom, int n2)
{
    int i = blockIdx.x * blockDim.x + threadIdx.x;
    if (i < n2) { m[i] = -INFINITY; denom[i] = 0.0f; }
}

__global__ void zero_kernel(float* p, size_t n)
{
    size_t i = (size_t)blockIdx.x * blockDim.x + threadIdx.x;
    if (i < n) p[i] = 0.0f;
}

// ---------------------------------------------------------------------------
// Pass 1: per-edge logits + segment max.  One wave per edge; lanes split the
// 64-wide head dot (2 floats each), wave reduction via shfl_xor.
// ---------------------------------------------------------------------------
__global__ void edge_logits_kernel(const float* __restrict__ q,
                                   const float* __restrict__ k,
                                   const long long* __restrict__ ei,
                                   float* __restrict__ logits,
                                   float* __restrict__ m)
{
    int wid  = (int)(((size_t)blockIdx.x * blockDim.x + threadIdx.x) >> 5);
    if (wid >= NE) return;
    int lane = threadIdx.x & 31;
    long long src = ei[wid];
    long long dst = ei[(size_t)NE + wid];

    float acc[2];
#pragma unroll
    for (int h = 0; h < 2; ++h) {
        float2 qv = *(const float2*)(q + (size_t)dst * FD + h * DH + lane * 2);
        float2 kv = *(const float2*)(k + (size_t)src * FD + h * DH + lane * 2);
        float p = qv.x * kv.x + qv.y * kv.y;
#pragma unroll
        for (int off = 16; off > 0; off >>= 1) p += __shfl_xor(p, off, 32);
        acc[h] = p * 0.125f;   // 1/sqrt(64)
    }
    if (lane == 0) {
        logits[(size_t)wid * 2 + 0] = acc[0];
        logits[(size_t)wid * 2 + 1] = acc[1];
        atomicMaxF(&m[(size_t)dst * 2 + 0], acc[0]);
        atomicMaxF(&m[(size_t)dst * 2 + 1], acc[1]);
    }
}

// ---------------------------------------------------------------------------
// Pass 2: e = exp(l - m[dst]); denom[dst] += e; num[dst] += e * v[src].
// One wave per edge; 4 fp32 atomic adds per lane into num (L2-resident).
// ---------------------------------------------------------------------------
__global__ void edge_accum_kernel(const float* __restrict__ vbuf,
                                  const long long* __restrict__ ei,
                                  const float* __restrict__ logits,
                                  const float* __restrict__ m,
                                  float* __restrict__ denom,
                                  float* __restrict__ num)
{
    int wid  = (int)(((size_t)blockIdx.x * blockDim.x + threadIdx.x) >> 5);
    if (wid >= NE) return;
    int lane = threadIdx.x & 31;
    long long src = ei[wid];
    long long dst = ei[(size_t)NE + wid];

    float e0 = __expf(logits[(size_t)wid * 2 + 0] - m[(size_t)dst * 2 + 0]);
    float e1 = __expf(logits[(size_t)wid * 2 + 1] - m[(size_t)dst * 2 + 1]);

    if (lane == 0) {
        atomicAddF(&denom[(size_t)dst * 2 + 0], e0);
        atomicAddF(&denom[(size_t)dst * 2 + 1], e1);
    }

    float2 v0 = *(const float2*)(vbuf + (size_t)src * FD + lane * 2);
    float2 v1 = *(const float2*)(vbuf + (size_t)src * FD + DH + lane * 2);
    float* nb = num + (size_t)dst * FD;
    atomicAddF(nb + lane * 2 + 0,      e0 * v0.x);
    atomicAddF(nb + lane * 2 + 1,      e0 * v0.y);
    atomicAddF(nb + DH + lane * 2 + 0, e1 * v1.x);
    atomicAddF(nb + DH + lane * 2 + 1, e1 * v1.y);
}

// xh = cvt_f16( skip(h) + num/max(denom,1e-16) [, ELU] )  — fused, no f32
// round trip of the layer output.
__global__ void finalize_cvt_kernel(const float* __restrict__ h,
                                    const float* __restrict__ num,
                                    const float* __restrict__ denom,
                                    _Float16* __restrict__ xh,
                                    int do_elu, size_t total)
{
    size_t i = (size_t)blockIdx.x * blockDim.x + threadIdx.x;
    if (i >= total) return;
    size_t node = i >> 7;
    int    hd   = (int)((i & 127) >> 6);
    float  den  = denom[node * 2 + hd];
    float  val  = h[i] + num[i] / fmaxf(den, 1e-16f);
    if (do_elu) val = (val > 0.0f) ? val : (__expf(val) - 1.0f);
    xh[i] = (_Float16)val;
}

// ---------------------------------------------------------------------------
extern "C" void kernel_launch(void* const* d_in, const int* in_sizes, int n_in,
                              void* d_out, int out_size, void* d_ws, size_t ws_size,
                              hipStream_t stream)
{
    const float*     x   = (const float*)d_in[0];
    const long long* ei  = (const long long*)d_in[1];   // int64 edge_index [2, E]
    const float* Wq1 = (const float*)d_in[2];  const float* bq1 = (const float*)d_in[3];
    const float* Wk1 = (const float*)d_in[4];  const float* bk1 = (const float*)d_in[5];
    const float* Wv1 = (const float*)d_in[6];  const float* bv1 = (const float*)d_in[7];
    const float* Ws1 = (const float*)d_in[8];  const float* bs1 = (const float*)d_in[9];
    const float* Wq2 = (const float*)d_in[10]; const float* bq2 = (const float*)d_in[11];
    const float* Wk2 = (const float*)d_in[12]; const float* bk2 = (const float*)d_in[13];
    const float* Wv2 = (const float*)d_in[14]; const float* bv2 = (const float*)d_in[15];
    const float* Ws2 = (const float*)d_in[16]; const float* bs2 = (const float*)d_in[17];
    const float* Wc  = (const float*)d_in[18]; const float* bc  = (const float*)d_in[19];
    float* out = (float*)d_out;

    // -------- workspace layout (256B aligned slots) --------
    char*  ws  = (char*)d_ws;
    size_t off = 0;
    auto alloc = [&](size_t bytes) -> char* {
        char* p = ws + off;
        off = (off + bytes + 255) & ~(size_t)255;
        return p;
    };
    const size_t NF = (size_t)NN * FD;
    float*     qbuf   = (float*)alloc(NF * 4);          // q  (reused as num)
    float*     kbuf   = (float*)alloc(NF * 4);          // k
    float*     vbuf   = (float*)alloc(NF * 4);          // v
    float*     hbuf   = (float*)alloc(NF * 4);          // skip output
    _Float16*  xh     = (_Float16*)alloc(NF * 2);       // f16 activations (A operand)
    float*     logits = (float*)alloc((size_t)NE * 2 * 4);
    float*     mbuf   = (float*)alloc((size_t)NN * 2 * 4);
    float*     denom  = (float*)alloc((size_t)NN * 2 * 4);
    _Float16*  wpack  = (_Float16*)alloc((size_t)9 * 16384 * 2);  // 8 full + Wc

    float* numbuf = qbuf;   // q dead after logits pass -> reuse as num accumulator

    const int STRIPES = NN / 16;                   // 6250
    const int GEMM_BLK = (STRIPES + 7) / 8;        // 782 (8 stripes / block)
    const int EW_BLK = (NE * 32) / 256;            // 200000 blocks, 1 wave/edge
    const int EL_BLK = (int)(NF / 256);            // 50000 elementwise blocks
    const int MD_BLK = (2 * NN + 255) / 256;

    // -------- prep: pack weights, f16 copy of x --------
    pack_weights_kernel<<<dim3(4, 9), 256, 0, stream>>>(
        Wq1, Wk1, Wv1, Ws1, Wq2, Wk2, Wv2, Ws2, Wc, wpack);
    cvt_f16_kernel<<<EL_BLK, 256, 0, stream>>>(x, xh, NF);

    // -------- two TransformerConv layers --------
    for (int layer = 0; layer < 2; ++layer) {
        const _Float16* wp4 = wpack + (size_t)layer * 4 * 16384;
        const float *b0, *b1, *b2, *b3;
        if (layer == 0) { b0 = bq1; b1 = bk1; b2 = bv1; b3 = bs1; }
        else            { b0 = bq2; b1 = bk2; b2 = bv2; b3 = bs2; }

        gemm_qkvs_kernel<<<dim3(GEMM_BLK, 4), 256, 0, stream>>>(
            xh, wp4, b0, b1, b2, b3, qbuf, kbuf, vbuf, hbuf);

        init_mdenom_kernel<<<MD_BLK, 256, 0, stream>>>(mbuf, denom, 2 * NN);

        edge_logits_kernel<<<EW_BLK, 256, 0, stream>>>(qbuf, kbuf, ei, logits, mbuf);

        zero_kernel<<<EL_BLK, 256, 0, stream>>>(numbuf, NF);  // q -> num

        edge_accum_kernel<<<EW_BLK, 256, 0, stream>>>(vbuf, ei, logits, mbuf,
                                                      denom, numbuf);

        finalize_cvt_kernel<<<EL_BLK, 256, 0, stream>>>(hbuf, numbuf, denom, xh,
                                                        (layer == 0) ? 1 : 0, NF);
    }

    // -------- classifier head: [100000,128] @ [128,40] + bc --------
    gemm_head_kernel<<<GEMM_BLK, 256, 0, stream>>>(xh, wpack + (size_t)8 * 16384,
                                                   bc, out);
}
